// MyGAT_20212116095193
// MI455X (gfx1250) — compile-verified, matrix operand order checked by reading, not run on hardware
//
#include <hip/hip_runtime.h>
#include <hip/hip_bf16.h>
#include <stdint.h>

// ---------------------------------------------------------------------------
// Problem constants (from reference)
// ---------------------------------------------------------------------------
#define BB      8
#define NN      512
#define EMBED   256
#define MROWS   (BB * NN)          // 4096
#define NEGV    (-1.0e12f)

typedef __attribute__((ext_vector_type(16))) _Float16 v16h;
typedef __attribute__((ext_vector_type(8)))  float    v8f;
typedef __attribute__((ext_vector_type(4)))  uint32_t u32x4;
typedef __attribute__((ext_vector_type(8)))  uint32_t u32x8;

// ---------------------------------------------------------------------------
// K0: tiny LUTs.  edot[layer][16] = edge_emb[t] . we ;  fh/ft[layer][4]
// luts layout (float): [0..31]=edot(l0,l1)  [32..39]=fh(l0,l1) [40..47]=ft
// ---------------------------------------------------------------------------
__global__ void build_luts(const float* __restrict__ edge_emb,
                           const float* __restrict__ flag_emb,
                           const float* __restrict__ w1,
                           const float* __restrict__ w2,
                           float* __restrict__ luts) {
  int tid = threadIdx.x;                 // 64 threads
  if (tid < 32) {                        // edge LUTs
    int layer = tid >> 4, t = tid & 15;
    const float* w = (layer ? w2 : w1) + 2 * EMBED + 64;   // we = w[576:608]
    float s = 0.f;
    for (int e = 0; e < 32; ++e) s += edge_emb[t * 32 + e] * w[e];
    luts[layer * 16 + t] = s;
  } else if (tid < 48) {                 // flag-head LUTs
    int idx = tid - 32, layer = idx >> 2, q = idx & 3;
    const float* w = (layer ? w2 : w1) + 2 * EMBED;        // wfh = w[512:544]
    float s = 0.f;
    for (int e = 0; e < 32; ++e) s += flag_emb[q * 32 + e] * w[e];
    luts[32 + layer * 4 + q] = s;
  } else {                               // flag-tail LUTs
    int idx = tid - 48, layer = idx >> 2, q = idx & 3;
    const float* w = (layer ? w2 : w1) + 2 * EMBED + 32;   // wft = w[544:576]
    float s = 0.f;
    for (int e = 0; e < 32; ++e) s += flag_emb[q * 32 + e] * w[e];
    luts[40 + layer * 4 + q] = s;
  }
}

// ---------------------------------------------------------------------------
// K1/K4: out16[m,d] = in[m,:] . W[d,:] + bias[d]   (W row-major 256x256)
// GATHER: rows come from node_emb[gather_idx[m]] (f32) else from in16 (f16).
// Block = 256 threads / 8 waves, one 16-row M tile per block; each wave owns
// a 32-column stripe (two 16x16 WMMA accumulators).
// ---------------------------------------------------------------------------
template <bool GATHER>
__global__ void transform_gemm(const float* __restrict__ nemb,
                               const _Float16* __restrict__ in16,
                               const int* __restrict__ gather_idx,
                               const float* __restrict__ W,
                               const float* __restrict__ bias,
                               _Float16* __restrict__ out16) {
  const int tid = threadIdx.x;
  const int m0  = blockIdx.x * 16;
  __shared__ _Float16 At[16][264];                 // padded 16x256 f16 A tile

  {                                                // stage A tile
    const int r  = tid >> 4;
    const int kb = (tid & 15) * 16;
    if (GATHER) {
      const int src  = gather_idx[m0 + r];
      const float* p = nemb + (size_t)src * EMBED + kb;
#pragma unroll
      for (int q = 0; q < 16; ++q) At[r][kb + q] = (_Float16)p[q];
    } else {
      const _Float16* p = in16 + (size_t)(m0 + r) * EMBED + kb;
#pragma unroll
      for (int q = 0; q < 16; ++q) At[r][kb + q] = p[q];
    }
  }
  __syncthreads();

  const int lane  = tid & 31;
  const int wave  = tid >> 5;
  const int row   = lane & 15;                     // A row / B column / C col
  const int kbase = (lane < 16) ? 0 : 8;           // half-wave K split
  const int d0    = wave * 32;

  v8f acc0 = {}, acc1 = {};

  for (int k0 = 0; k0 < EMBED; k0 += 32) {
    v16h a, bA, bBv;
    const float* w0 = W + (size_t)(d0 + row) * EMBED + k0;        // col d0+row
    const float* w1p = W + (size_t)(d0 + 16 + row) * EMBED + k0;  // col d0+16+row
#pragma unroll
    for (int v = 0; v < 8; ++v) {
      const int kp = (v < 4) ? (kbase + 2 * v) : (16 + kbase + 2 * (v - 4));
      a[2 * v]     = At[row][k0 + kp];
      a[2 * v + 1] = At[row][k0 + kp + 1];
      bA[2 * v]     = (_Float16)w0[kp];
      bA[2 * v + 1] = (_Float16)w0[kp + 1];
      bBv[2 * v]     = (_Float16)w1p[kp];
      bBv[2 * v + 1] = (_Float16)w1p[kp + 1];
    }
    acc0 = __builtin_amdgcn_wmma_f32_16x16x32_f16(false, a, false, bA,
                                                  (short)0, acc0, false, false);
    acc1 = __builtin_amdgcn_wmma_f32_16x16x32_f16(false, a, false, bBv,
                                                  (short)0, acc1, false, false);
  }

  const int n  = lane & 15;
  const int mh = (lane < 16) ? 0 : 8;
#pragma unroll
  for (int v = 0; v < 8; ++v) {
    const int m = m0 + mh + v;
    const int d = d0 + n;
    out16[(size_t)m * EMBED + d]      = (_Float16)(acc0[v] + bias[d]);
    out16[(size_t)m * EMBED + d + 16] = (_Float16)(acc1[v] + bias[d + 16]);
  }
}

// ---------------------------------------------------------------------------
// K2/K5: rh[m] = h[m,:].wh ; rt[m] = h[m,:].wt   (wh=w[0:256], wt=w[256:512])
// ---------------------------------------------------------------------------
__global__ void row_scores(const _Float16* __restrict__ h16,
                           const float* __restrict__ w,
                           float* __restrict__ rh, float* __restrict__ rt) {
  const int m = blockIdx.x * blockDim.x + threadIdx.x;   // 4096 rows
  const _Float16* p = h16 + (size_t)m * EMBED;
  float s0 = 0.f, s1 = 0.f;
#pragma unroll 8
  for (int d = 0; d < EMBED; ++d) {
    const float hv = (float)p[d];
    s0 += hv * w[d];
    s1 += hv * w[EMBED + d];
  }
  rh[m] = s0;
  rt[m] = s1;
}

// ---------------------------------------------------------------------------
// K3: fused GAT attention for one (batch b, 16-wide j tile).
//   logit[i,jj] = rh[i]+rt[j]+fh[flag_i]+ft[flag_j]+edot[etype]+bias
//   -> leaky -> mask -> softmax over i (denominator folded into epilogue)
//   out[j,d] = (1/sum_j) * sum_i exp(logit[i,j]-max_j) * h[i,d]   via WMMA
// Phase-2 h chunks (32x256 f16) are DMA'd into LDS with the CDNA5 Tensor
// Data Mover (tensor_load_to_lds, TENSORcnt), issued by wave 0 only.
// FINAL=false: out = f16 sigmoid(out) -> t16 ;  FINAL=true: f32 -> d_out
// LDS: phase-1 logits (32KB) region is reused as the phase-2 h chunk.
// ---------------------------------------------------------------------------
template <bool FINAL>
__global__ void gat_attention(const _Float16* __restrict__ h16,
                              const float* __restrict__ rh,
                              const float* __restrict__ rt,
                              const int* __restrict__ adj,
                              const int* __restrict__ etype,
                              const int* __restrict__ flags,
                              const float* __restrict__ edot,
                              const float* __restrict__ fhl,
                              const float* __restrict__ ftl,
                              const float* __restrict__ bias_ptr,
                              _Float16* __restrict__ out16,
                              float* __restrict__ outf) {
  const int tid = threadIdx.x;
  const int b   = blockIdx.y;
  const int j0  = blockIdx.x * 16;

  __shared__ __align__(16) unsigned char smem0[NN * 16 * sizeof(float)]; // 32KB
  float    (*logits)[16] = (float (*)[16])smem0;       // phase 1
  _Float16 (*hch)[256]   = (_Float16 (*)[256])smem0;   // phase 2 (32x256 f16)
  __shared__ _Float16 attT[16][520];                   // exp(logit-max), f16
  __shared__ float red[16][17];
  __shared__ float cmax[16], csum[16], colc[16];

  const float bias = bias_ptr[0];
  const int*  flg  = flags + b * NN;

  if (tid < 16) {
    const int j = j0 + tid;
    colc[tid] = rt[b * NN + j] + ftl[flg[j]];
  }
  __syncthreads();

  // ---- phase 1: logits -------------------------------------------------
  for (int i = tid; i < NN; i += 256) {
    const float rv = rh[b * NN + i] + fhl[flg[i]];
    const int* arow = adj   + ((size_t)(b * NN + i)) * NN + j0;
    const int* erow = etype + ((size_t)(b * NN + i)) * NN + j0;
#pragma unroll
    for (int jj = 0; jj < 16; ++jj) {
      float v = rv + colc[jj] + edot[erow[jj]] + bias;
      v = (v > 0.f) ? v : 0.1f * v;                 // leaky relu
      if (arow[jj] == 0) v = NEGV;                  // adjacency mask
      logits[i][jj] = v;
    }
  }
  __syncthreads();

  // column max over i
  {
    const int jj = tid & 15, seg = tid >> 4;
    float m = -3.4e38f;
    for (int s = 0; s < 32; ++s) m = fmaxf(m, logits[seg * 32 + s][jj]);
    red[jj][seg] = m;
  }
  __syncthreads();
  if (tid < 16) {
    float m = red[tid][0];
    for (int s = 1; s < 16; ++s) m = fmaxf(m, red[tid][s]);
    cmax[tid] = m;
  }
  __syncthreads();

  // exp + partial sums; store unnormalized att^T as f16
  {
    const int jj = tid & 15, seg = tid >> 4;
    const float mx = cmax[jj];
    float s = 0.f;
    for (int t = 0; t < 32; ++t) {
      const int i = seg * 32 + t;
      const float e = __expf(logits[i][jj] - mx);
      s += e;
      attT[jj][i] = (_Float16)e;
    }
    red[jj][seg] = s;
  }
  __syncthreads();
  if (tid < 16) {
    float s = 0.f;
    for (int t = 0; t < 16; ++t) s += red[tid][t];
    csum[tid] = 1.0f / s;
  }
  __syncthreads();   // logits dead from here; smem0 becomes the h chunk

  // ---- phase 2: out(16x256) = attT(16x512) x h(512x256) via WMMA -------
  const int lane  = tid & 31;
  const int wave  = tid >> 5;
  const int row   = lane & 15;
  const int kbase = (lane < 16) ? 0 : 8;
  const int d0    = wave * 32;
  const uint32_t lds_off = (uint32_t)(size_t)&hch[0][0];  // LDS byte offset
  v8f acc0 = {}, acc1 = {};

  for (int k0 = 0; k0 < NN; k0 += 32) {
    if (wave == 0) {
      // TDM: DMA the 32x256 f16 h chunk (16KB) global -> LDS in one shot.
      const uint64_t ga =
          (uint64_t)(size_t)(h16 + ((size_t)(b * NN + k0)) * EMBED);
      u32x4 g0;
      g0[0] = 1u;                                   // count=1, user descriptor
      g0[1] = lds_off;                              // lds_addr
      g0[2] = (uint32_t)ga;                         // global_addr[31:0]
      g0[3] = (uint32_t)(ga >> 32) | (2u << 30);    // global_addr[56:32]|type=2
      u32x8 g1;
      g1[0] = (1u << 16);        // workgroup_mask=0, data_size=1 (2 bytes)
      g1[1] = (256u << 16);      // tensor_dim0 = 256 elements
      g1[2] = (32u << 16);       // tensor_dim1 = 32 rows
      g1[3] = (256u << 16);      // tile_dim0 = 256
      g1[4] = 32u;               // tile_dim1 = 32, tile_dim2 = 0
      g1[5] = 256u;              // tensor_dim0_stride[31:0] = 256
      g1[6] = 0u;                // stride hi / dim1_stride lo
      g1[7] = 0u;                // dim1_stride hi
      asm volatile("tensor_load_to_lds %0, %1" :: "s"(g0), "s"(g1) : "memory");
      __builtin_amdgcn_s_wait_tensorcnt(0);         // TENSORcnt -> 0
    } else if (wave == 1 && k0 + 32 < NN) {
      // warm L2/WGP$ for the next chunk (global_prefetch_b8)
      __builtin_prefetch(
          h16 + ((size_t)(b * NN + k0 + 32)) * EMBED + (lane << 4), 0, 1);
    }
    __syncthreads();

    v16h a, b0f, b1f;
#pragma unroll
    for (int v = 0; v < 8; ++v) {
      const int kp = (v < 4) ? (kbase + 2 * v) : (16 + kbase + 2 * (v - 4));
      a[2 * v]     = attT[row][k0 + kp];
      a[2 * v + 1] = attT[row][k0 + kp + 1];
      b0f[2 * v]     = hch[kp][d0 + row];
      b0f[2 * v + 1] = hch[kp + 1][d0 + row];
      b1f[2 * v]     = hch[kp][d0 + 16 + row];
      b1f[2 * v + 1] = hch[kp + 1][d0 + 16 + row];
    }
    acc0 = __builtin_amdgcn_wmma_f32_16x16x32_f16(false, a, false, b0f,
                                                  (short)0, acc0, false, false);
    acc1 = __builtin_amdgcn_wmma_f32_16x16x32_f16(false, a, false, b1f,
                                                  (short)0, acc1, false, false);
    __syncthreads();                       // chunk consumed before re-stage
  }

  // ---- epilogue: fold softmax denominator, activation, store ------------
  const int n  = lane & 15;
  const int mh = (lane < 16) ? 0 : 8;
#pragma unroll
  for (int v = 0; v < 8; ++v) {
    const int m   = mh + v;                // local j row of C tile
    const int j   = j0 + m;
    const float inv = csum[m];
    const float v0  = acc0[v] * inv;
    const float v1  = acc1[v] * inv;
    const size_t o  = ((size_t)(b * NN + j)) * EMBED + d0 + n;
    if (FINAL) {
      outf[o]      = v0;
      outf[o + 16] = v1;
    } else {
      out16[o]      = (_Float16)(1.f / (1.f + __expf(-v0)));
      out16[o + 16] = (_Float16)(1.f / (1.f + __expf(-v1)));
    }
  }
}

// ---------------------------------------------------------------------------
// Host-side orchestration
// ---------------------------------------------------------------------------
extern "C" void kernel_launch(void* const* d_in, const int* in_sizes, int n_in,
                              void* d_out, int out_size, void* d_ws,
                              size_t ws_size, hipStream_t stream) {
  (void)in_sizes; (void)n_in; (void)out_size; (void)ws_size;

  const int*   adj      = (const int*)d_in[0];
  const int*   heads    = (const int*)d_in[1];
  const int*   flags    = (const int*)d_in[2];
  const int*   etype    = (const int*)d_in[3];
  const float* node_emb = (const float*)d_in[4];
  const float* edge_emb = (const float*)d_in[5];
  const float* flag_emb = (const float*)d_in[6];
  const float* t1_w     = (const float*)d_in[7];
  const float* t1_b     = (const float*)d_in[8];
  const float* w1       = (const float*)d_in[9];
  const float* b1       = (const float*)d_in[10];
  const float* t2_w     = (const float*)d_in[11];
  const float* t2_b     = (const float*)d_in[12];
  const float* w2       = (const float*)d_in[13];
  const float* b2       = (const float*)d_in[14];

  char* ws = (char*)d_ws;
  _Float16* h16 = (_Float16*)(ws);                        // 2 MB
  _Float16* t16 = (_Float16*)(ws + ((size_t)2 << 20));    // 2 MB
  _Float16* g16 = (_Float16*)(ws + ((size_t)4 << 20));    // 2 MB
  float* rh1  = (float*)(ws + ((size_t)6 << 20));
  float* rt1  = rh1 + MROWS;
  float* rh2  = rt1 + MROWS;
  float* rt2  = rh2 + MROWS;
  float* luts = rt2 + MROWS;                              // 48 floats
  float* outf = (float*)d_out;

  build_luts<<<1, 64, 0, stream>>>(edge_emb, flag_emb, w1, w2, luts);

  // layer 1: h = node_emb[heads] @ t1_w.T + t1_b (f16), scores, attention
  transform_gemm<true><<<MROWS / 16, 256, 0, stream>>>(node_emb, nullptr,
                                                       heads, t1_w, t1_b, h16);
  row_scores<<<MROWS / 256, 256, 0, stream>>>(h16, w1, rh1, rt1);
  gat_attention<false><<<dim3(NN / 16, BB), 256, 0, stream>>>(
      h16, rh1, rt1, adj, etype, flags, luts + 0, luts + 32, luts + 40, b1,
      t16, nullptr);

  // layer 2: h2 = sigmoid(t) @ t2_w.T + t2_b, scores, attention -> d_out
  transform_gemm<false><<<MROWS / 16, 256, 0, stream>>>(nullptr, t16, nullptr,
                                                        t2_w, t2_b, g16);
  row_scores<<<MROWS / 256, 256, 0, stream>>>(g16, w2, rh2, rt2);
  gat_attention<true><<<dim3(NN / 16, BB), 256, 0, stream>>>(
      g16, rh2, rt2, adj, etype, flags, luts + 16, luts + 36, luts + 44, b2,
      nullptr, outf);
}
